// EllEssTeeEmm_27161373180447
// MI455X (gfx1250) — 16-bit and 8-bit matrix operand order repaired and checked by reading, compiled, not run on hardware
//
#include <hip/hip_runtime.h>


typedef __attribute__((ext_vector_type(16))) __bf16 v16bf;
typedef __attribute__((ext_vector_type(8)))  float  v8f;
typedef __attribute__((ext_vector_type(8)))  int    v8i;
typedef __attribute__((ext_vector_type(4)))  int    v4i;

#define LSTM_B 32
#define LSTM_T 2048
#define LSTM_H 256
#define LSTM_G 1024
#define LSTM_M (LSTM_B * LSTM_T)

#define H_ROW (LSTM_H + 16)

__device__ __forceinline__ float sigf(float x) {
  return 1.0f / (1.0f + __expf(-x));
}

__device__ __forceinline__ unsigned int f32_to_e4m3_sw(float x) {
  unsigned int u = __float_as_uint(x);
  unsigned int s = (u >> 24) & 0x80u;
  float ax = fabsf(x);
  if (!(ax >= 0.0009765625f)) return s;
  if (ax > 448.0f) ax = 448.0f;
  int e;
  float m = frexpf(ax, &e);
  int E = e + 6;
  unsigned int bits;
  if (E >= 1) {
    int q = (int)rintf(m * 16.0f);
    if (q == 16) { q = 8; ++E; }
    if (E > 15) { E = 15; q = 14; }
    bits = s | ((unsigned int)E << 3) | ((unsigned int)q & 7u);
  } else {
    int q = (int)rintf(ax * 512.0f);
    if (q > 7) q = 7;
    bits = s | (unsigned int)q;
  }
  return bits;
}

__device__ __forceinline__ unsigned int pk_fp8_pair(float a, float b) {
#if __has_builtin(__builtin_amdgcn_cvt_pk_fp8_f32)
  return (unsigned int)__builtin_amdgcn_cvt_pk_fp8_f32(a, b, 0, false) & 0xffffu;
#else
  return f32_to_e4m3_sw(a) | (f32_to_e4m3_sw(b) << 8);
#endif
}

template <int K>
__global__ void __launch_bounds__(256)
lstm_xg_gemm(const float* __restrict__ X, const float* __restrict__ W,
             const float* __restrict__ bias, float* __restrict__ out) {
  const int lane = threadIdx.x & 31;
  const int wave = threadIdx.x >> 5;
  const int r    = lane & 15;
  const int half = lane >> 4;

  const int mBase = blockIdx.x * 16;
  const int nBase = (blockIdx.y * 8 + wave) * 16;
  const int n     = nBase + r;

  v8f acc;
  {
    const float bv = bias[n];
#pragma unroll
    for (int v = 0; v < 8; ++v) acc[v] = bv;
  }

  const float* xrow = X + (size_t)(mBase + r) * K;
  const float* wrow = W + (size_t)n * K;

  for (int k0 = 0; k0 < K; k0 += 32) {
    union { v16bf v; __bf16 e[16]; } a, b;
    const float4 xa0 = *(const float4*)(xrow + k0 + half * 8);
    const float4 xa1 = *(const float4*)(xrow + k0 + half * 8 + 4);
    const float4 xb0 = *(const float4*)(xrow + k0 + 16 + half * 8);
    const float4 xb1 = *(const float4*)(xrow + k0 + 16 + half * 8 + 4);
    a.e[0]  = (__bf16)xa0.x; a.e[1]  = (__bf16)xa0.y; a.e[2]  = (__bf16)xa0.z; a.e[3]  = (__bf16)xa0.w;
    a.e[4]  = (__bf16)xa1.x; a.e[5]  = (__bf16)xa1.y; a.e[6]  = (__bf16)xa1.z; a.e[7]  = (__bf16)xa1.w;
    a.e[8]  = (__bf16)xb0.x; a.e[9]  = (__bf16)xb0.y; a.e[10] = (__bf16)xb0.z; a.e[11] = (__bf16)xb0.w;
    a.e[12] = (__bf16)xb1.x; a.e[13] = (__bf16)xb1.y; a.e[14] = (__bf16)xb1.z; a.e[15] = (__bf16)xb1.w;

    const float* wp = wrow + k0 + half * 8;
#pragma unroll
    for (int q = 0; q < 4; ++q) {
      const float4 w4 = *(const float4*)(wp + (q & 1) * 4 + (q >> 1) * 16);
      b.e[q * 4 + 0] = (__bf16)w4.x; b.e[q * 4 + 1] = (__bf16)w4.y;
      b.e[q * 4 + 2] = (__bf16)w4.z; b.e[q * 4 + 3] = (__bf16)w4.w;
    }

    acc = __builtin_amdgcn_wmma_f32_16x16x32_bf16(false, a.v, false, b.v,
                                                  (short)0, acc, false, false);
  }

  float* orow = out + (size_t)(mBase + half * 8) * LSTM_G + n;
#pragma unroll
  for (int v = 0; v < 8; ++v) orow[(size_t)v * LSTM_G] = acc[v];
}

__global__ void __launch_bounds__(512, 1)
lstm_scan2(const float* __restrict__ xgF, const float* __restrict__ WhhF,
           const float* __restrict__ xgR, const float* __restrict__ WhhR,
           float* __restrict__ hout) {
  const int dir  = blockIdx.x;
  const float* xg = dir ? xgR : xgF;
  const float* W  = dir ? WhhR : WhhF;
  const int hoff  = dir ? LSTM_H : 0;

  const int tid  = threadIdx.x;
  const int lane = tid & 31;
  const int wave = tid >> 5;
  const int r    = lane & 15;
  const int half = lane >> 4;
  const int hcol = wave * 16 + r;

  __shared__ __align__(16) unsigned char wpk[16 * 4 * 4 * 1024];
  __shared__ __align__(16) unsigned char hl[2][LSTM_B][H_ROW];

  for (int it = tid; it < 1024 * 64; it += 512) {
    const int n  = it >> 6;
    const int k4 = it & 63;
    const float4 w4 = *(const float4*)(W + (size_t)n * LSTM_H + k4 * 4);
    const unsigned int pk =
        pk_fp8_pair(w4.x, w4.y) | (pk_fp8_pair(w4.z, w4.w) << 16);
    const int g  = n >> 8, wv = (n >> 4) & 15, rr = n & 15;
    const int K  = k4 * 4;
    const int k  = K >> 6;
    const int rem = K & 63;
    const int chunk = (rem & 32) ? 1 : 0;
    const int hf = (rem & 31) >> 4;
    const int j  = rem & 15;
    const int L  = hf * 16 + rr;
    *(unsigned int*)&wpk[(((wv * 4 + g) * 4 + k) * 1024) + chunk * 512 + L * 16 + j] = pk;
  }
  for (int i = tid; i < LSTM_B * H_ROW; i += 512) (&hl[0][0][0])[i] = 0;

  v8f cst[2];
#pragma unroll
  for (int mt = 0; mt < 2; ++mt)
#pragma unroll
    for (int v = 0; v < 8; ++v) cst[mt][v] = 0.0f;

  __syncthreads();

  int p = 0;
  for (int t = 0; t < LSTM_T; ++t) {
    const int tc = dir ? (LSTM_T - 1 - t) : t;

    v8f acc[4][2];
#pragma unroll
    for (int g = 0; g < 4; ++g) {
      const int n = g * 256 + hcol;
#pragma unroll
      for (int mt = 0; mt < 2; ++mt) {
#pragma unroll
        for (int v = 0; v < 8; ++v) {
          const int b = mt * 16 + half * 8 + v;
          acc[g][mt][v] = xg[((size_t)b * LSTM_T + tc) * LSTM_G + n];
        }
      }
    }
    if (t + 1 < LSTM_T) {
      const int tn = dir ? (LSTM_T - 2 - t) : (t + 1);
#pragma unroll
      for (int g = 0; g < 4; ++g) {
        const int n = g * 256 + hcol;
        __builtin_prefetch(&xg[((size_t)(half * 8) * LSTM_T + tn) * LSTM_G + n], 0, 0);
        __builtin_prefetch(&xg[((size_t)(16 + half * 8) * LSTM_T + tn) * LSTM_G + n], 0, 0);
      }
    }

#pragma unroll
    for (int k = 0; k < 4; ++k) {
      union { v8i v; unsigned long long q[4]; } a[2];
#pragma unroll
      for (int mt = 0; mt < 2; ++mt) {
        const unsigned char* hr = &hl[p][mt * 16 + r][k * 64 + half * 16];
        a[mt].q[0] = *(const unsigned long long*)(hr);
        a[mt].q[1] = *(const unsigned long long*)(hr + 8);
        a[mt].q[2] = *(const unsigned long long*)(hr + 32);
        a[mt].q[3] = *(const unsigned long long*)(hr + 40);
      }
#pragma unroll
      for (int g = 0; g < 4; ++g) {
        union { v8i v; v4i h[2]; } bf;
        const unsigned char* bp = &wpk[(((wave * 4 + g) * 4 + k) * 1024) + lane * 16];
        bf.h[0] = *(const v4i*)(bp);
        bf.h[1] = *(const v4i*)(bp + 512);
        acc[g][0] = __builtin_amdgcn_wmma_f32_16x16x64_fp8_fp8(
            a[0].v, bf.v, (short)0, acc[g][0], false, false);
        acc[g][1] = __builtin_amdgcn_wmma_f32_16x16x64_fp8_fp8(
            a[1].v, bf.v, (short)0, acc[g][1], false, false);
      }
    }

#pragma unroll
    for (int mt = 0; mt < 2; ++mt) {
      float hv[8];
#pragma unroll
      for (int v = 0; v < 8; ++v) {
        const float ig = sigf(acc[0][mt][v]);
        const float fg = sigf(acc[1][mt][v]);
        const float gg = tanhf(acc[2][mt][v]);
        const float og = sigf(acc[3][mt][v]);
        const float c  = fg * cst[mt][v] + ig * gg;
        cst[mt][v] = c;
        hv[v] = og * tanhf(c);
        const int b = mt * 16 + half * 8 + v;
        hout[((size_t)b * LSTM_T + tc) * (2 * LSTM_H) + hoff + hcol] = hv[v];
      }
#pragma unroll
      for (int v = 0; v < 8; v += 2) {
        const unsigned int pk = pk_fp8_pair(hv[v], hv[v + 1]);
        const int b = mt * 16 + half * 8 + v;
        hl[1 - p][b][hcol]     = (unsigned char)(pk & 0xffu);
        hl[1 - p][b + 1][hcol] = (unsigned char)(pk >> 8);
      }
    }

    __syncthreads();
    p ^= 1;
  }
}

extern "C" void kernel_launch(void* const* d_in, const int* in_sizes, int n_in,
                              void* d_out, int out_size, void* d_ws, size_t ws_size,
                              hipStream_t stream) {
  (void)in_sizes; (void)n_in; (void)out_size; (void)ws_size;
  const float* x      = (const float*)d_in[0];
  const float* Wih_f0 = (const float*)d_in[1];
  const float* Whh_f0 = (const float*)d_in[2];
  const float* b_f0   = (const float*)d_in[3];
  const float* Wih_r0 = (const float*)d_in[4];
  const float* Whh_r0 = (const float*)d_in[5];
  const float* b_r0   = (const float*)d_in[6];
  const float* Wih_f1 = (const float*)d_in[7];
  const float* Whh_f1 = (const float*)d_in[8];
  const float* b_f1   = (const float*)d_in[9];
  const float* Wih_r1 = (const float*)d_in[10];
  const float* Whh_r1 = (const float*)d_in[11];
  const float* b_r1   = (const float*)d_in[12];

  float* ws  = (float*)d_ws;
  float* xgF = ws;
  float* xgR = xgF + (size_t)LSTM_M * LSTM_G;
  float* h0  = xgR + (size_t)LSTM_M * LSTM_G;
  float* out = (float*)d_out;

  const dim3 gBlk(256);
  const dim3 gGrd(LSTM_M / 16, LSTM_G / (16 * 8));

  lstm_xg_gemm<256><<<gGrd, gBlk, 0, stream>>>(x, Wih_f0, b_f0, xgF);
  lstm_xg_gemm<256><<<gGrd, gBlk, 0, stream>>>(x, Wih_r0, b_r0, xgR);
  lstm_scan2<<<dim3(2), dim3(512), 0, stream>>>(xgF, Whh_f0, xgR, Whh_r0, h0);

  lstm_xg_gemm<512><<<gGrd, gBlk, 0, stream>>>(h0, Wih_f1, b_f1, xgF);
  lstm_xg_gemm<512><<<gGrd, gBlk, 0, stream>>>(h0, Wih_r1, b_r1, xgR);
  lstm_scan2<<<dim3(2), dim3(512), 0, stream>>>(xgF, Whh_f1, xgR, Whh_r1, out);
}
